// LocalRNN_49340584296877
// MI455X (gfx1250) — compile-verified, matrix operand order checked by reading, not run on hardware
//
#include <hip/hip_runtime.h>

// ---- plain clang ext-vector types (POD, union-safe) ----
typedef __attribute__((ext_vector_type(16))) __bf16        v16bf;
typedef __attribute__((ext_vector_type(8)))  float         v8f;
typedef __attribute__((ext_vector_type(4)))  float         f4;
typedef __attribute__((ext_vector_type(4)))  unsigned int  u4;

union FragU { u4 q[2]; v16bf v; };

// Build a bf16 A/B fragment (16x16x32 wmma, 16-bit operand layout) from fp32
// memory. Per ISA 7.12.2: lane holds K = {c..c+7} (elems 0-7) and
// {c+16..c+23} (elems 8-15), where p points at element c of the row.
__device__ __forceinline__ v16bf load_frag_f32(const float* __restrict__ p) {
    f4 a0 = *(const f4*)(p);
    f4 a1 = *(const f4*)(p + 4);
    f4 a2 = *(const f4*)(p + 16);
    f4 a3 = *(const f4*)(p + 20);
    v16bf v;
    v[0] = (__bf16)a0.x; v[1] = (__bf16)a0.y; v[2]  = (__bf16)a0.z; v[3]  = (__bf16)a0.w;
    v[4] = (__bf16)a1.x; v[5] = (__bf16)a1.y; v[6]  = (__bf16)a1.z; v[7]  = (__bf16)a1.w;
    v[8] = (__bf16)a2.x; v[9] = (__bf16)a2.y; v[10] = (__bf16)a2.z; v[11] = (__bf16)a2.w;
    v[12] = (__bf16)a3.x; v[13] = (__bf16)a3.y; v[14] = (__bf16)a3.z; v[15] = (__bf16)a3.w;
    return v;
}

// Same fragment but sourced from bf16 LDS (two ds_load_b128).
__device__ __forceinline__ v16bf load_frag_lds(const __bf16* p) {
    FragU f;
    f.q[0] = *(const u4*)(p);        // K = c .. c+7
    f.q[1] = *(const u4*)(p + 16);   // K = c+16 .. c+23
    return f.v;
}

// CDNA5 has native v_tanh_f32 (TRANS pipe); use it when clang exposes it.
__device__ __forceinline__ float fast_tanh(float x) {
#if __has_builtin(__builtin_amdgcn_tanhf)
    return __builtin_amdgcn_tanhf(x);
#else
    float e = __expf(2.0f * x);
    return (e - 1.0f) / (e + 1.0f);
#endif
}
__device__ __forceinline__ float sigf(float x) {
    return 0.5f * fast_tanh(0.5f * x) + 0.5f;   // 1 trans op instead of exp+rcp
}

// ---------------------------------------------------------------------------
// Kernel 1: gih[t, 0:512] = w_ih @ x[t] + b_ih     (t = 0..16383)
// One block = one 16-row M tile; 8 waves x 4 N-tiles = 512 gate columns.
// Block 0 additionally emits a bf16 copy of w_hh for kernel 2's async stage.
// ---------------------------------------------------------------------------
__global__ __launch_bounds__(256) void lrnn_gates_ih(
    const float* __restrict__ x,     // [16384,128]
    const float* __restrict__ w_ih,  // [512,128]
    const float* __restrict__ b_ih,  // [512]
    const float* __restrict__ w_hh,  // [512,128]
    float* __restrict__ gih,         // [16384,512]
    __bf16* __restrict__ whh_bf16)   // [512,128]
{
    const int tid   = threadIdx.x;
    const int wv    = tid >> 5;        // wave 0..7
    const int ln    = tid & 31;
    const int lhalf = ln >> 4;         // 0 / 1
    const int l16   = ln & 15;
    const int mt    = blockIdx.x;      // 0..1023

    if (mt == 0) {                     // one-time w_hh fp32 -> bf16 for kernel 2
        for (int idx = tid; idx < 512 * 128; idx += 256)
            whh_bf16[idx] = (__bf16)w_hh[idx];
    }

    const int row = mt * 16 + l16;     // A-fragment row for this lane
    v8f acc[4] = {};

    #pragma unroll
    for (int ks = 0; ks < 4; ++ks) {
        const int c = ks * 32 + lhalf * 8;
        const v16bf a = load_frag_f32(x + row * 128 + c);
        #pragma unroll
        for (int i = 0; i < 4; ++i) {
            const int g = (wv * 4 + i) * 16 + l16;     // B-fragment column (gate)
            const v16bf b = load_frag_f32(w_ih + g * 128 + c);
            acc[i] = __builtin_amdgcn_wmma_f32_16x16x32_bf16(
                false, a, false, b, (short)0, acc[i], false, false);
        }
    }

    #pragma unroll
    for (int i = 0; i < 4; ++i) {
        const int n    = (wv * 4 + i) * 16 + l16;
        const float bi = b_ih[n];
        #pragma unroll
        for (int r = 0; r < 8; ++r) {
            const int m = r + lhalf * 8;               // C layout: M = r (+8 for hi half)
            gih[(mt * 16 + m) * 512 + n] = acc[i][r] + bi;
        }
    }
}

// ---------------------------------------------------------------------------
// Kernel 2: 16-step LSTM recurrence. 4 waves own a group of 16 tokens
// (same batch, consecutive s). Wave w computes gate columns [w*128,(w+1)*128)
// of gates = Hbf16 @ w_hh^T via WMMA; 128 lanes/group do the cell update.
// Two groups per 256-thread block share the bf16 w_hh copy in LDS, which is
// staged with global_load_async_to_lds_b128 (ASYNCcnt).
// ---------------------------------------------------------------------------
__global__ __launch_bounds__(256) void lrnn_recurrent(
    const __bf16* __restrict__ whh_bf16, // [512,128] bf16 (from kernel 1)
    const float* __restrict__ b_ih,      // [512]
    const float* __restrict__ b_hh,      // [512]
    const float* __restrict__ gih,       // [16384,512]
    float* __restrict__ out)             // [16384,128]
{
    __shared__ __bf16 whh_lds[512 * 128];     // 128 KB, row-major [g][k]
    __shared__ float  gbuf[2][16][512];       //  64 KB, gates per group
    __shared__ __bf16 hbuf[2][16 * 128];      //   8 KB, h per group, row-major

    const int tid = threadIdx.x;

    // ---- async-stage w_hh bf16 into LDS (128 KB = 32 sweeps x 256 x 16B) ----
    {
        const unsigned lds_base = (unsigned)(unsigned long long)(void*)whh_lds;
        #pragma unroll 4
        for (int it = 0; it < 32; ++it) {
            const unsigned off = (unsigned)(it * 256 + tid) * 16u;
            const unsigned lds_addr = lds_base + off;
            const void* g = (const char*)whh_bf16 + off;
            asm volatile("global_load_async_to_lds_b128 %0, %1, off"
                         :: "v"(lds_addr), "v"(g) : "memory");
        }
        asm volatile("s_wait_asynccnt 0" ::: "memory");
    }
    __syncthreads();

    const int wv    = tid >> 5;
    const int grp   = wv >> 2;                 // 0 / 1 (== tid>>7)
    const int wloc  = wv & 3;                  // wave-in-group = gate group
    const int ln    = tid & 31;
    const int lhalf = ln >> 4;
    const int l16   = ln & 15;

    const int gid = blockIdx.x * 2 + grp;      // token group 0..1023
    const int b   = gid >> 7;                  // batch
    const int s0  = (gid & 127) * 16;          // first token s in group

    // Elementwise mapping: lane d = tid&127 owns hidden unit d, all 16 tokens.
    const int d = tid & 127;
    const float bi_i = b_ih[d], bi_f = b_ih[128 + d], bi_g = b_ih[256 + d], bi_o = b_ih[384 + d];
    const float bh_i = b_hh[d], bh_f = b_hh[128 + d], bh_g = b_hh[256 + d], bh_o = b_hh[384 + d];

    float c_st[16];
    #pragma unroll
    for (int m = 0; m < 16; ++m) c_st[m] = 0.0f;

    const long grow = (long)b * 2048;

    for (int j = 0; j < 16; ++j) {
        // Windows overlap: step j introduces exactly one new gih row (t = s0+j).
        // Prefetch it (2 KB) while the matmul runs.  (global_prefetch_b8)
        if (wloc == 0 && ln < 16)
            __builtin_prefetch((const char*)(gih + (grow + s0 + j) * 512) + ln * 128, 0, 0);

        // ---- recurrent matmul: gates_slice = Hbf @ Whh_slice^T ----
        v8f acc[8] = {};
        if (j != 0) {                          // h == 0 at step 0: skip 32 WMMAs
            #pragma unroll
            for (int ks = 0; ks < 4; ++ks) {
                const int c = ks * 32 + lhalf * 8;
                const v16bf a = load_frag_lds(&hbuf[grp][l16 * 128 + c]);
                #pragma unroll
                for (int i = 0; i < 8; ++i) {
                    const int g = wloc * 128 + i * 16 + l16;
                    const v16bf bb = load_frag_lds(&whh_lds[g * 128 + c]);
                    acc[i] = __builtin_amdgcn_wmma_f32_16x16x32_bf16(
                        false, a, false, bb, (short)0, acc[i], false, false);
                }
            }
        }
        // spill gate slice to LDS (C layout -> [token][gate])
        #pragma unroll
        for (int i = 0; i < 8; ++i) {
            const int n = wloc * 128 + i * 16 + l16;
            #pragma unroll
            for (int r = 0; r < 8; ++r)
                gbuf[grp][r + lhalf * 8][n] = acc[i][r];
        }
        __syncthreads();

        // ---- elementwise LSTM cell ----
        #pragma unroll
        for (int m = 0; m < 16; ++m) {
            const int t = s0 + m - 15 + j;     // absolute input position
            float gxi, gxf, gxg, gxo;
            if (t >= 0) {
                const float* gr = gih + (grow + t) * 512;
                gxi = gr[d]; gxf = gr[128 + d]; gxg = gr[256 + d]; gxo = gr[384 + d];
            } else {                           // zero-padded row: xg == b_ih
                gxi = bi_i; gxf = bi_f; gxg = bi_g; gxo = bi_o;
            }
            const float gi = gbuf[grp][m][d]       + gxi + bh_i;
            const float gf = gbuf[grp][m][128 + d] + gxf + bh_f;
            const float gg = gbuf[grp][m][256 + d] + gxg + bh_g;
            const float go = gbuf[grp][m][384 + d] + gxo + bh_o;

            const float cn = sigf(gf) * c_st[m] + sigf(gi) * fast_tanh(gg);
            c_st[m] = cn;
            const float hn = sigf(go) * fast_tanh(cn);

            if (j == 15)
                out[(grow + s0 + m) * 128 + d] = hn;
            else
                hbuf[grp][m * 128 + d] = (__bf16)hn;
        }
        __syncthreads();
    }
}

extern "C" void kernel_launch(void* const* d_in, const int* in_sizes, int n_in,
                              void* d_out, int out_size, void* d_ws, size_t ws_size,
                              hipStream_t stream) {
    const float* x    = (const float*)d_in[0];   // [8,2048,128]
    const float* w_ih = (const float*)d_in[1];   // [512,128]
    const float* w_hh = (const float*)d_in[2];   // [512,128]
    const float* b_ih = (const float*)d_in[3];   // [512]
    const float* b_hh = (const float*)d_in[4];   // [512]
    float* out = (float*)d_out;                  // [8,2048,128] fp32
    float* gih = (float*)d_ws;                   // 16384*512 fp32 = 33.5 MB scratch
    __bf16* whh_bf16 = (__bf16*)(gih + 16384 * 512);  // +128 KB bf16 w_hh

    lrnn_gates_ih<<<1024, 256, 0, stream>>>(x, w_ih, b_ih, w_hh, gih, whh_bf16);
    lrnn_recurrent<<<512, 256, 0, stream>>>(whh_bf16, b_ih, b_hh, gih, out);
}
